// StandardMoELayer_45999099740752
// MI455X (gfx1250) — compile-verified, matrix operand order checked by reading, not run on hardware
//
#include <hip/hip_runtime.h>

// ---------------------------------------------------------------------------
// StandardMoELayer on MI455X (gfx1250): bf16 WMMA GEMMs (dense GEMMs staged
// through LDS by the Tensor Data Mover), flash attention, and a gathered
// (winning-expert-only) MoE grouped GEMM.
// ---------------------------------------------------------------------------

#define DEV __device__ __forceinline__

constexpr int Tt = 1024;          // sequence length
constexpr int BB = 8;             // batch
constexpr int C  = 1024;          // model dim
constexpr int C3 = 3 * C;
constexpr int N  = Tt * BB;       // tokens = 8192
constexpr int H  = 16;
constexpr int D  = C / H;         // 64
constexpr int E  = 8;
constexpr int MAXTILES = N / 16 + E;   // 520 (worst-case padded row tiles)
constexpr int NPAD     = N + 16 * E;   // 8320 gathered-row capacity

typedef __attribute__((ext_vector_type(16))) __bf16 v16bf;
typedef __attribute__((ext_vector_type(8)))  float  v8f;
typedef __attribute__((ext_vector_type(4)))  unsigned int u32x4;
typedef __attribute__((ext_vector_type(8)))  int i32x8;
typedef __attribute__((ext_vector_type(4)))  int i32x4;

union V16u { v16bf v; __bf16 h[16]; uint4 q[2]; };

// Load a 16x32 bf16 operand tile in the WMMA A/B register layout:
// lane&15 selects the row (A) / column-of-B row-in-memory, lane>>4 selects the
// 8-element K group; halves 0-7 = K+{kg*8..}, halves 8-15 = K+16+{kg*8..}.
// Works on both global and LDS (generic) pointers.
DEV v16bf load_tile16(const __bf16* __restrict__ base, int stride) {
  const int lane = threadIdx.x & 31;
  const __bf16* p = base + (size_t)(lane & 15) * stride + (lane >> 4) * 8;
  V16u t;
  t.q[0] = *reinterpret_cast<const uint4*>(p);
  t.q[1] = *reinterpret_cast<const uint4*>(p + 16);
  return t.v;
}

DEV v8f wmma_bf16(v16bf a, v16bf b, v8f c) {
  return __builtin_amdgcn_wmma_f32_16x16x32_bf16(false, a, false, b,
                                                 (short)0, c, false, false);
}

DEV float gelu_exact(float x) {
  return 0.5f * x * (1.0f + erff(x * 0.70710678118654752440f));
}

// ------------------------- Tensor Data Mover helpers -----------------------

// Flat LDS pointer -> 32-bit LDS byte address (flat addr[31:0] is the LDS
// offset per the aperture mapping).
DEV unsigned lds_addr_of(const void* p) {
  return (unsigned)(uintptr_t)p;
}

// 2-D TDM load: tile of tile_k (contiguous, dim0) x tile_rows (dim1) bf16
// elements from global memory (row stride = row_stride elements) into LDS at
// `lds` (rows stored densely, row stride = tile_k elements).
// Descriptor packing per CDNA5 ISA 8.3/8.4 (group0 128b, group1 256b).
// This toolchain exposes the 6-arg builtin:
//   (u32x4 g0, i32x8 g1, i32x4 g2, i32x4 g3, i32x8 g4, i32 cpol)
DEV void tdm_load_2d(unsigned lds, const __bf16* gptr, int tile_k,
                     int tile_rows, int row_stride) {
  union { u32x4 v; unsigned w[4]; } g0;
  union { i32x8 v; unsigned w[8]; } g1;
  const unsigned long long ga = (unsigned long long)(uintptr_t)gptr;
  g0.w[0] = 1u;                                     // count=1, user descriptor
  g0.w[1] = lds;                                    // lds_addr (bytes)
  g0.w[2] = (unsigned)(ga & 0xffffffffu);           // global_addr[31:0]
  g0.w[3] = (unsigned)((ga >> 32) & 0x01ffffffu)    // global_addr[56:32]
          | (2u << 30);                             // type = 2 ("image")
  g1.w[0] = (1u << 16);                             // data_size=1 -> 2 bytes
  g1.w[1] = ((unsigned)tile_k & 0xffffu) << 16;     // tensor_dim0[15:0]
  g1.w[2] = ((unsigned)tile_rows & 0xffffu) << 16;  // tensor_dim1[15:0]
  g1.w[3] = ((unsigned)tile_k & 0xffffu) << 16;     // tile_dim0
  g1.w[4] = (unsigned)tile_rows & 0xffffu;          // tile_dim1 (tile_dim2=0)
  g1.w[5] = (unsigned)row_stride;                   // tensor_dim0_stride[31:0]
  g1.w[6] = 0;                                      // stride[47:32], dim1_stride lo
  g1.w[7] = 0;
  i32x4 z4 = {0, 0, 0, 0};
  i32x8 z8 = {0, 0, 0, 0, 0, 0, 0, 0};
  __builtin_amdgcn_tensor_load_to_lds(g0.v, g1.v, z4, z4, z8, 0);
}

// 16 rows x 64 cols accumulation straight from global (MoE grouped GEMMs).
DEV void gemm_core_16x64(const __bf16* __restrict__ A,
                         const __bf16* __restrict__ W, int K, v8f acc[4]) {
  for (int k0 = 0; k0 < K; k0 += 32) {
    v16bf a  = load_tile16(A + k0, K);
    v16bf b0 = load_tile16(W + (size_t)0  * K + k0, K);
    v16bf b1 = load_tile16(W + (size_t)16 * K + k0, K);
    v16bf b2 = load_tile16(W + (size_t)32 * K + k0, K);
    v16bf b3 = load_tile16(W + (size_t)48 * K + k0, K);
    acc[0] = wmma_bf16(a, b0, acc[0]);
    acc[1] = wmma_bf16(a, b1, acc[1]);
    acc[2] = wmma_bf16(a, b2, acc[2]);
    acc[3] = wmma_bf16(a, b3, acc[3]);
  }
}

// --------------------------- elementwise helpers ---------------------------

__global__ void cvt_bf16_kernel(const float* __restrict__ src,
                                __bf16* __restrict__ dst, int n) {
  int i = blockIdx.x * blockDim.x + threadIdx.x;
  if (i < n) dst[i] = (__bf16)src[i];
}

// vT[(b*H+h)*D*T + d*T + s] = qkv_v[(s*B+b)*3C + 2C + h*D + d]
__global__ void transpose_v_kernel(const __bf16* __restrict__ qkv,
                                   __bf16* __restrict__ vT) {
  int idx  = blockIdx.x * blockDim.x + threadIdx.x;   // over N*C
  int s    = idx % Tt;
  int rest = idx / Tt;
  int d    = rest % D;
  int bh   = rest / D;
  int b = bh / H, h = bh % H;
  vT[idx] = qkv[((size_t)s * BB + b) * C3 + 2 * C + h * D + d];
}

// ---------------- dense GEMM + bias (TDM-staged, double-buffered) ----------
// Block = 4 waves = 64 rows x 64 cols. Wave 0 drives the TDM: one 64x64 A
// panel + one 64x64 B panel per stage into LDS (double buffered, 32 KB);
// all 4 waves consume them with ds_load_b128 -> v_wmma. Loads for a k-chunk
// are batched before the WMMA chain so one s_wait_dscnt covers them.

__global__ __launch_bounds__(128)
void gemm_tdm_kernel(const __bf16* __restrict__ A, const __bf16* __restrict__ W,
                     const float* __restrict__ bias, float* __restrict__ outF,
                     __bf16* __restrict__ outB, int M, int Nc, int K) {
  __shared__ __bf16 abuf[2][64 * 64];
  __shared__ __bf16 bbuf[2][64 * 64];
  const int lane = threadIdx.x & 31;
  const int wave = threadIdx.x >> 5;
  const int m0 = blockIdx.y * 64;
  const int n0 = blockIdx.x * 64;
  const __bf16* Ab = A + (size_t)m0 * K;
  const __bf16* Wb = W + (size_t)n0 * K;

  if (wave == 0) {
    tdm_load_2d(lds_addr_of(abuf[0]), Ab, 64, 64, K);
    tdm_load_2d(lds_addr_of(bbuf[0]), Wb, 64, 64, K);
    __builtin_amdgcn_s_wait_tensorcnt(0);
  }
  __syncthreads();

  v8f acc[4] = {};
  const int nstages = K / 64;
  for (int i = 0; i < nstages; ++i) {
    const int cur = i & 1, nxt = cur ^ 1;
    if (wave == 0 && i + 1 < nstages) {
      tdm_load_2d(lds_addr_of(abuf[nxt]), Ab + (i + 1) * 64, 64, 64, K);
      tdm_load_2d(lds_addr_of(bbuf[nxt]), Wb + (i + 1) * 64, 64, 64, K);
    }
    for (int kc = 0; kc < 2; ++kc) {
      v16bf a = load_tile16(abuf[cur] + wave * 16 * 64 + kc * 32, 64);
      v16bf b[4];
      for (int nt = 0; nt < 4; ++nt)
        b[nt] = load_tile16(bbuf[cur] + nt * 16 * 64 + kc * 32, 64);
      for (int nt = 0; nt < 4; ++nt)
        acc[nt] = wmma_bf16(a, b[nt], acc[nt]);
    }
    if (wave == 0) __builtin_amdgcn_s_wait_tensorcnt(0);
    __syncthreads();   // next buffer ready + current buffer free for overwrite
  }

  const int n = lane & 15, hi = lane >> 4;
  const int mw = m0 + wave * 16;
  if (mw >= M) return;
  for (int nt = 0; nt < 4; ++nt) {
    const int col = n0 + nt * 16 + n;
    const float bv = bias ? bias[col] : 0.0f;
    for (int r = 0; r < 8; ++r) {
      const int m = mw + r + hi * 8;
      const float v = acc[nt][r] + bv;
      if (outF) outF[(size_t)m * Nc + col] = v;
      if (outB) outB[(size_t)m * Nc + col] = (__bf16)v;
    }
  }
}

// ----------------------------- flash attention -----------------------------

__global__ __launch_bounds__(128)
void attn_kernel(const __bf16* __restrict__ qkv, const __bf16* __restrict__ vT,
                 __bf16* __restrict__ ctx) {
  __shared__ float pbuf[4][16 * 32];
  const int lane = threadIdx.x & 31;
  const int wave = threadIdx.x >> 5;
  const int bh = blockIdx.y;
  const int b = bh / H, h = bh % H;
  const int t0 = (blockIdx.x * 4 + wave) * 16;
  const int rowStride = BB * C3;
  const __bf16* qbase = qkv + (size_t)b * C3 + h * D;
  const __bf16* kbase = qkv + (size_t)b * C3 + C + h * D;
  const __bf16* vTb   = vT + (size_t)bh * D * Tt;

  v16bf qa0 = load_tile16(qbase + (size_t)t0 * rowStride + 0,  rowStride);
  v16bf qa1 = load_tile16(qbase + (size_t)t0 * rowStride + 32, rowStride);

  float mr[8], lr[8];
  v8f acc[4] = {};
  for (int r = 0; r < 8; ++r) { mr[r] = -1e30f; lr[r] = 0.0f; }
  const int n = lane & 15, hi = lane >> 4;
  float* lp = pbuf[wave];

  for (int s0 = 0; s0 < Tt; s0 += 32) {
    v8f S0 = {}, S1 = {};
    {
      v16bf kb0 = load_tile16(kbase + (size_t)s0 * rowStride + 0,  rowStride);
      v16bf kb1 = load_tile16(kbase + (size_t)s0 * rowStride + 32, rowStride);
      v16bf kc0 = load_tile16(kbase + (size_t)(s0 + 16) * rowStride + 0,  rowStride);
      v16bf kc1 = load_tile16(kbase + (size_t)(s0 + 16) * rowStride + 32, rowStride);
      S0 = wmma_bf16(qa0, kb0, S0); S0 = wmma_bf16(qa1, kb1, S0);
      S1 = wmma_bf16(qa0, kc0, S1); S1 = wmma_bf16(qa1, kc1, S1);
    }
    float rm[8], fac[8], rs[8];
    for (int r = 0; r < 8; ++r) {
      float x0 = S0[r] * 0.125f, x1 = S1[r] * 0.125f;  // 1/sqrt(64)
      S0[r] = x0; S1[r] = x1;
      rm[r] = fmaxf(x0, x1);
    }
    for (int msk = 1; msk < 16; msk <<= 1)
      for (int r = 0; r < 8; ++r)
        rm[r] = fmaxf(rm[r], __shfl_xor(rm[r], msk, 32));
    for (int r = 0; r < 8; ++r) {
      float mn = fmaxf(mr[r], rm[r]);
      fac[r] = __expf(mr[r] - mn);
      mr[r] = mn;
      float p0 = __expf(S0[r] - mn), p1 = __expf(S1[r] - mn);
      S0[r] = p0; S1[r] = p1;
      rs[r] = p0 + p1;
    }
    for (int msk = 1; msk < 16; msk <<= 1)
      for (int r = 0; r < 8; ++r)
        rs[r] += __shfl_xor(rs[r], msk, 32);
    for (int r = 0; r < 8; ++r) lr[r] = lr[r] * fac[r] + rs[r];
    for (int nt = 0; nt < 4; ++nt)
      for (int r = 0; r < 8; ++r) acc[nt][r] *= fac[r];

    // C-layout -> LDS -> A-layout transpose of P, f32 -> bf16.
    for (int r = 0; r < 8; ++r) {
      lp[(r + hi * 8) * 32 + n]      = S0[r];
      lp[(r + hi * 8) * 32 + 16 + n] = S1[r];
    }
    __syncthreads();
    V16u pa;
    {
      const int m = lane & 15, kg = lane >> 4;
      const float* src = lp + m * 32 + kg * 8;
      for (int j = 0; j < 8; ++j) pa.h[j]     = (__bf16)src[j];
      for (int j = 0; j < 8; ++j) pa.h[8 + j] = (__bf16)src[16 + j];
    }
    __syncthreads();
    for (int nt = 0; nt < 4; ++nt) {
      v16bf vb = load_tile16(vTb + (size_t)(nt * 16) * Tt + s0, Tt);
      acc[nt] = wmma_bf16(pa.v, vb, acc[nt]);
    }
  }

  for (int r = 0; r < 8; ++r) {
    const float inv = 1.0f / lr[r];
    const int t = t0 + r + hi * 8;
    const size_t base = ((size_t)t * BB + b) * C + h * D;
    for (int nt = 0; nt < 4; ++nt)
      ctx[base + nt * 16 + n] = (__bf16)(acc[nt][r] * inv);
  }
}

// --------------------------- residual + layernorm --------------------------

__global__ __launch_bounds__(256)
void add_ln_kernel(const float* __restrict__ a, const float* __restrict__ b,
                   const float* __restrict__ g, const float* __restrict__ be,
                   float* __restrict__ outF, __bf16* __restrict__ outB) {
  __shared__ float ws1[8], ws2[8];
  const int t = blockIdx.x;
  const float* ra = a + (size_t)t * C;
  const float* rb = b + (size_t)t * C;
  float vals[4], s = 0.0f, s2 = 0.0f;
  for (int i = 0; i < 4; ++i) {
    int c = threadIdx.x + i * 256;
    float v = ra[c] + rb[c];
    vals[i] = v; s += v; s2 += v * v;
  }
  for (int msk = 16; msk; msk >>= 1) {
    s  += __shfl_xor(s,  msk, 32);
    s2 += __shfl_xor(s2, msk, 32);
  }
  const int wave = threadIdx.x >> 5, lane = threadIdx.x & 31;
  if (lane == 0) { ws1[wave] = s; ws2[wave] = s2; }
  __syncthreads();
  if (wave == 0) {
    float a1 = (lane < 8) ? ws1[lane] : 0.0f;
    float a2 = (lane < 8) ? ws2[lane] : 0.0f;
    for (int msk = 4; msk; msk >>= 1) {
      a1 += __shfl_xor(a1, msk, 32);
      a2 += __shfl_xor(a2, msk, 32);
    }
    if (lane == 0) { ws1[0] = a1; ws2[0] = a2; }
  }
  __syncthreads();
  const float mean = ws1[0] * (1.0f / C);
  const float var  = ws2[0] * (1.0f / C) - mean * mean;
  const float rstd = rsqrtf(var + 1e-5f);
  for (int i = 0; i < 4; ++i) {
    int c = threadIdx.x + i * 256;
    float vv = (vals[i] - mean) * rstd * g[c] + be[c];
    if (outF) outF[(size_t)t * C + c] = vv;
    if (outB) outB[(size_t)t * C + c] = (__bf16)vv;
  }
}

// --------------------------------- router ----------------------------------

__global__ __launch_bounds__(256)
void router_kernel(const float* __restrict__ hb, const float* __restrict__ rw,
                   const float* __restrict__ rb, int* __restrict__ assign) {
  const int wave = threadIdx.x >> 5, lane = threadIdx.x & 31;
  const int t = blockIdx.x * 8 + wave;
  float acc[E] = {};
  const float* xr = hb + (size_t)t * C;
  for (int c = lane; c < C; c += 32) {
    float x = xr[c];
    for (int e = 0; e < E; ++e) acc[e] += x * rw[e * C + c];
  }
  for (int msk = 16; msk; msk >>= 1)
    for (int e = 0; e < E; ++e) acc[e] += __shfl_xor(acc[e], msk, 32);
  if (lane == 0) {
    float lg[E];
    for (int e = 0; e < E; ++e) lg[e] = acc[e] + rb[e];
    int i1 = 0;
    for (int e = 1; e < E; ++e) if (lg[e] > lg[i1]) i1 = e;   // top_k tie: lower idx
    int i2 = -1;
    for (int e = 0; e < E; ++e) {
      if (e == i1) continue;
      if (i2 < 0 || lg[e] > lg[i2]) i2 = e;
    }
    assign[t] = (i1 > i2) ? i1 : i2;   // last-write-wins == max index of top-2
  }
}

// --------------------------- MoE compaction chain --------------------------

__global__ void moe_init_kernel(int* __restrict__ cnt, int* __restrict__ row_tok) {
  int i = blockIdx.x * blockDim.x + threadIdx.x;
  if (i < E) cnt[i] = 0;
  if (i < NPAD) row_tok[i] = -1;
}

__global__ void moe_count_kernel(const int* __restrict__ assign,
                                 int* __restrict__ cnt, int* __restrict__ slot) {
  int t = blockIdx.x * blockDim.x + threadIdx.x;
  if (t < N) slot[t] = atomicAdd(&cnt[assign[t]], 1);
}

__global__ void moe_scan_kernel(const int* __restrict__ cnt, int* __restrict__ off,
                                int* __restrict__ tile_e) {
  int o = 0;
  for (int e = 0; e < E; ++e) {
    off[e] = o;
    int tiles = (cnt[e] + 15) >> 4;
    for (int i = 0; i < tiles; ++i) tile_e[(o >> 4) + i] = e;
    o += tiles * 16;
  }
  off[E] = o;
  for (int tv = o >> 4; tv < MAXTILES; ++tv) tile_e[tv] = -1;
}

__global__ void moe_zeropad_kernel(const int* __restrict__ cnt,
                                   const int* __restrict__ off,
                                   __bf16* __restrict__ xg) {
  const int e = blockIdx.x >> 4, j = blockIdx.x & 15;
  const int c = cnt[e];
  const int padded = (c + 15) & ~15;
  if (c + j >= padded) return;
  const size_t row = (size_t)off[e] + c + j;
  for (int k = threadIdx.x; k < C; k += 256) xg[row * C + k] = (__bf16)0.0f;
}

__global__ void moe_gather_kernel(const int* __restrict__ assign,
                                  const int* __restrict__ slot,
                                  const int* __restrict__ off,
                                  const __bf16* __restrict__ hb,
                                  __bf16* __restrict__ xg,
                                  int* __restrict__ row_tok) {
  const int t = blockIdx.x;
  const int row = off[assign[t]] + slot[t];
  if (threadIdx.x == 0) row_tok[row] = t;
  for (int c = threadIdx.x; c < C; c += 256)
    xg[(size_t)row * C + c] = hb[(size_t)t * C + c];
}

// ------------------------- MoE grouped expert GEMMs ------------------------

__global__ __launch_bounds__(128)
void moe_gemm1_kernel(const __bf16* __restrict__ xg, const __bf16* __restrict__ w1,
                      const float* __restrict__ b1, const int* __restrict__ tile_e,
                      __bf16* __restrict__ hid) {
  const int lane = threadIdx.x & 31;
  const int wave = threadIdx.x >> 5;
  const int rt = blockIdx.y * 4 + wave;
  if (rt >= MAXTILES) return;
  const int e = tile_e[rt];
  if (e < 0) return;
  const int n0 = blockIdx.x * 64;
  v8f acc[4] = {};
  gemm_core_16x64(xg + (size_t)rt * 16 * C,
                  w1 + (size_t)e * C * C + (size_t)n0 * C, C, acc);
  const int n = lane & 15, hi = lane >> 4;
  for (int nt = 0; nt < 4; ++nt) {
    const int col = n0 + nt * 16 + n;
    const float bv = b1[e * C + col];
    for (int r = 0; r < 8; ++r) {
      const float v = gelu_exact(acc[nt][r] + bv);
      hid[((size_t)rt * 16 + r + hi * 8) * C + col] = (__bf16)v;
    }
  }
}

__global__ __launch_bounds__(128)
void moe_gemm2_kernel(const __bf16* __restrict__ hid, const __bf16* __restrict__ w2,
                      const float* __restrict__ b2, const int* __restrict__ tile_e,
                      const int* __restrict__ row_tok, float* __restrict__ moe) {
  const int lane = threadIdx.x & 31;
  const int wave = threadIdx.x >> 5;
  const int rt = blockIdx.y * 4 + wave;
  if (rt >= MAXTILES) return;
  const int e = tile_e[rt];
  if (e < 0) return;
  const int n0 = blockIdx.x * 64;
  v8f acc[4] = {};
  gemm_core_16x64(hid + (size_t)rt * 16 * C,
                  w2 + (size_t)e * C * C + (size_t)n0 * C, C, acc);
  const int n = lane & 15, hi = lane >> 4;
  for (int r = 0; r < 8; ++r) {
    const int tok = row_tok[rt * 16 + r + hi * 8];
    if (tok < 0) continue;
    for (int nt = 0; nt < 4; ++nt) {
      const int col = n0 + nt * 16 + n;
      moe[(size_t)tok * C + col] = acc[nt][r] + b2[e * C + col];
    }
  }
}

// --------------------------------- launch ----------------------------------

extern "C" void kernel_launch(void* const* d_in, const int* in_sizes, int n_in,
                              void* d_out, int out_size, void* d_ws, size_t ws_size,
                              hipStream_t stream) {
  const float* x     = (const float*)d_in[0];
  const float* in_w  = (const float*)d_in[1];
  const float* in_b  = (const float*)d_in[2];
  const float* out_w = (const float*)d_in[3];
  const float* out_b = (const float*)d_in[4];
  const float* ln1_g = (const float*)d_in[5];
  const float* ln1_b = (const float*)d_in[6];
  const float* rw    = (const float*)d_in[7];
  const float* rb    = (const float*)d_in[8];
  const float* w1    = (const float*)d_in[9];
  const float* b1    = (const float*)d_in[10];
  const float* w2    = (const float*)d_in[11];
  const float* b2    = (const float*)d_in[12];
  const float* ln2_g = (const float*)d_in[13];
  const float* ln2_b = (const float*)d_in[14];
  float* out = (float*)d_out;

  char* wsp = (char*)d_ws;
  auto carve = [&](size_t bytes) -> void* {
    void* p = (void*)wsp;
    wsp += (bytes + 255) & ~(size_t)255;
    return p;
  };

  __bf16* x_b    = (__bf16*)carve((size_t)N * C * 2);
  __bf16* wqkv_b = (__bf16*)carve((size_t)C3 * C * 2);
  __bf16* wout_b = (__bf16*)carve((size_t)C * C * 2);
  __bf16* w1_b   = (__bf16*)carve((size_t)E * C * C * 2);
  __bf16* w2_b   = (__bf16*)carve((size_t)E * C * C * 2);
  __bf16* qkv_b  = (__bf16*)carve((size_t)N * C3 * 2);
  __bf16* vT_b   = (__bf16*)carve((size_t)N * C * 2);
  __bf16* ctx_b  = (__bf16*)carve((size_t)N * C * 2);
  float*  attn_f = (float*)carve((size_t)N * C * 4);
  float*  h_f    = (float*)carve((size_t)N * C * 4);
  __bf16* h_b    = (__bf16*)carve((size_t)N * C * 2);
  float*  moe_f  = (float*)carve((size_t)N * C * 4);
  __bf16* xg_b   = (__bf16*)carve((size_t)NPAD * C * 2);
  __bf16* hid_b  = (__bf16*)carve((size_t)NPAD * C * 2);
  int* assign    = (int*)carve((size_t)N * 4);
  int* slot      = (int*)carve((size_t)N * 4);
  int* cnt       = (int*)carve((size_t)E * 4);
  int* off       = (int*)carve((size_t)(E + 1) * 4);
  int* tile_e    = (int*)carve((size_t)MAXTILES * 4);
  int* row_tok   = (int*)carve((size_t)NPAD * 4);

  cvt_bf16_kernel<<<(N * C + 255) / 256, 256, 0, stream>>>(x, x_b, N * C);
  cvt_bf16_kernel<<<(C3 * C + 255) / 256, 256, 0, stream>>>(in_w, wqkv_b, C3 * C);
  cvt_bf16_kernel<<<(C * C + 255) / 256, 256, 0, stream>>>(out_w, wout_b, C * C);
  cvt_bf16_kernel<<<(E * C * C + 255) / 256, 256, 0, stream>>>(w1, w1_b, E * C * C);
  cvt_bf16_kernel<<<(E * C * C + 255) / 256, 256, 0, stream>>>(w2, w2_b, E * C * C);

  gemm_tdm_kernel<<<dim3(C3 / 64, N / 64), 128, 0, stream>>>(
      x_b, wqkv_b, in_b, nullptr, qkv_b, N, C3, C);
  transpose_v_kernel<<<(N * C) / 256, 256, 0, stream>>>(qkv_b, vT_b);
  attn_kernel<<<dim3(Tt / 64, BB * H), 128, 0, stream>>>(qkv_b, vT_b, ctx_b);
  gemm_tdm_kernel<<<dim3(C / 64, N / 64), 128, 0, stream>>>(
      ctx_b, wout_b, out_b, attn_f, nullptr, N, C, C);
  add_ln_kernel<<<N, 256, 0, stream>>>(x, attn_f, ln1_g, ln1_b, h_f, h_b);

  router_kernel<<<N / 8, 256, 0, stream>>>(h_f, rw, rb, assign);
  moe_init_kernel<<<(NPAD + 255) / 256, 256, 0, stream>>>(cnt, row_tok);
  moe_count_kernel<<<N / 256, 256, 0, stream>>>(assign, cnt, slot);
  moe_scan_kernel<<<1, 1, 0, stream>>>(cnt, off, tile_e);
  moe_zeropad_kernel<<<E * 16, 256, 0, stream>>>(cnt, off, xg_b);
  moe_gather_kernel<<<N, 256, 0, stream>>>(assign, slot, off, h_b, xg_b, row_tok);
  moe_gemm1_kernel<<<dim3(C / 64, (MAXTILES + 3) / 4), 128, 0, stream>>>(
      xg_b, w1_b, b1, tile_e, hid_b);
  moe_gemm2_kernel<<<dim3(C / 64, (MAXTILES + 3) / 4), 128, 0, stream>>>(
      hid_b, w2_b, b2, tile_e, row_tok, moe_f);
  add_ln_kernel<<<N, 256, 0, stream>>>(h_f, moe_f, ln2_g, ln2_b, out, nullptr);

  (void)in_sizes; (void)n_in; (void)out_size; (void)ws_size;
}
